// CrossAttention_42786464203188
// MI455X (gfx1250) — matrix operand order repaired and checked by reading, not compiled, not run on hardware
//
#include <hip/hip_runtime.h>

typedef __attribute__((ext_vector_type(16))) __bf16 v16bf;
typedef __attribute__((ext_vector_type(8)))  float  v8f;
typedef __attribute__((ext_vector_type(4)))  float  f4;
typedef __attribute__((ext_vector_type(4)))  int    i4;
typedef __attribute__((ext_vector_type(2)))  int    i2;

union FragAB { v16bf v; i4 q[2]; };
union Pack8  { i4 q; unsigned short u[8]; };

__device__ __forceinline__ unsigned short f2bfh(float f) {
  return __builtin_bit_cast(unsigned short, (__bf16)f);   // hw cvt, RNE
}
__device__ __forceinline__ unsigned int pkbf(float lo, float hi) {
  return (unsigned int)f2bfh(lo) | ((unsigned int)f2bfh(hi) << 16);
}
__device__ __forceinline__ float bf2f(unsigned short h) {
  return __uint_as_float(((unsigned int)h) << 16);
}

// ---------------------------------------------------------------------------
// GEMM: O[b] (bf16, MxN) = W (f32, MxK, shared) @ X[b] (f32, KxN)
// Block tile 64(M) x 128(N), BK=32; 4 waves, each 32x64 (2x4 WMMA tiles).
// ---------------------------------------------------------------------------
__global__ __launch_bounds__(128) void k_gemm_f32w_bf16out(
    const float* __restrict__ W, const float* __restrict__ X,
    unsigned short* __restrict__ Out, int M, int K, int N,
    long xBatch, long oBatch)
{
  __shared__ unsigned short sW[64 * 32];    // [m][k] bf16
  __shared__ unsigned short sX[128 * 32];   // [n][k] bf16 (transposed on fill)
  const int b   = blockIdx.z;
  const int m0  = blockIdx.y * 64;
  const int n0  = blockIdx.x * 128;
  const float* Xb = X + (long)b * xBatch;
  unsigned short* Ob = Out + (long)b * oBatch;
  const int tid  = threadIdx.x;
  const int wid  = tid >> 5;
  const int lane = tid & 31;
  const int wm = (wid >> 1) * 32;
  const int wn = (wid & 1) * 64;
  const int l15 = lane & 15;
  const int akb = (lane < 16) ? 0 : 16;   // A: K lo 16 | hi 16
  const int bkb = (lane < 16) ? 0 : 16;   // B: K lo 16 | hi 16

  v8f zero = {};
  v8f acc[2][4];
  #pragma unroll
  for (int i = 0; i < 2; i++)
    #pragma unroll
    for (int j = 0; j < 4; j++) acc[i][j] = zero;

  for (int k0 = 0; k0 < K; k0 += 32) {
    // W tile: 64x32 f32 -> bf16.  512 float4 / 128 thr = 4 each.
    #pragma unroll
    for (int j = 0; j < 4; j++) {
      int idx = tid + j * 128;            // float4 index
      int r = idx >> 3, c4 = idx & 7;     // 8 float4 per 32-wide row
      f4 f = *(const f4*)&W[(long)(m0 + r) * K + (k0 + c4 * 4)];
      i2 p; p.x = (int)pkbf(f.x, f.y); p.y = (int)pkbf(f.z, f.w);
      *(i2*)&sW[r * 32 + c4 * 4] = p;
    }
    // X tile: 32(k) x 128(n) f32 -> bf16 transposed into [n][k].
    // unit = 2 k-rows x 4 n-cols; 512 units / 128 thr = 4 each.
    #pragma unroll
    for (int j = 0; j < 4; j++) {
      int u = tid + j * 128;
      int kk = (u & 15) * 2, nn = (u >> 4) * 4;
      f4 a = *(const f4*)&Xb[(long)(k0 + kk)     * N + (n0 + nn)];
      f4 c = *(const f4*)&Xb[(long)(k0 + kk + 1) * N + (n0 + nn)];
      *(unsigned int*)&sX[(nn + 0) * 32 + kk] = pkbf(a.x, c.x);
      *(unsigned int*)&sX[(nn + 1) * 32 + kk] = pkbf(a.y, c.y);
      *(unsigned int*)&sX[(nn + 2) * 32 + kk] = pkbf(a.z, c.z);
      *(unsigned int*)&sX[(nn + 3) * 32 + kk] = pkbf(a.w, c.w);
    }
    __syncthreads();
    FragAB a[2], bb[4];
    #pragma unroll
    for (int t = 0; t < 2; t++) {
      const unsigned short* ap = &sW[(wm + t * 16 + l15) * 32 + akb];
      a[t].q[0] = *(const i4*)ap;
      a[t].q[1] = *(const i4*)(ap + 8);
    }
    #pragma unroll
    for (int t = 0; t < 4; t++) {
      const unsigned short* bp = &sX[(wn + t * 16 + l15) * 32 + bkb];
      bb[t].q[0] = *(const i4*)bp;
      bb[t].q[1] = *(const i4*)(bp + 8);
    }
    #pragma unroll
    for (int tm = 0; tm < 2; tm++)
      #pragma unroll
      for (int tn = 0; tn < 4; tn++)
        acc[tm][tn] = __builtin_amdgcn_wmma_f32_16x16x32_bf16(
            false, a[tm].v, false, bb[tn].v, (short)0, acc[tm][tn], false, false);
    __syncthreads();
  }
  #pragma unroll
  for (int tm = 0; tm < 2; tm++) {
    int rb = m0 + wm + tm * 16 + ((lane < 16) ? 0 : 8);
    #pragma unroll
    for (int tn = 0; tn < 4; tn++) {
      int cc = n0 + wn + tn * 16 + l15;
      #pragma unroll
      for (int i = 0; i < 8; i++)
        Ob[(long)(rb + i) * N + cc] = f2bfh(acc[tm][tn][i]);
    }
  }
}

// ---------------------------------------------------------------------------
// GEMM: O[b] (f32, MxN) = A[b] (bf16, MxK) @ X[b] (bf16, KxN)
// Same tiling as above.
// ---------------------------------------------------------------------------
__global__ __launch_bounds__(128) void k_gemm_bf16_f32out(
    const unsigned short* __restrict__ A, const unsigned short* __restrict__ X,
    float* __restrict__ Out, int M, int K, int N,
    long aBatch, long xBatch, long oBatch)
{
  __shared__ unsigned short sW[64 * 32];
  __shared__ unsigned short sX[128 * 32];
  const int b  = blockIdx.z;
  const int m0 = blockIdx.y * 64;
  const int n0 = blockIdx.x * 128;
  const unsigned short* Ab = A + (long)b * aBatch;
  const unsigned short* Xb = X + (long)b * xBatch;
  float* Ob = Out + (long)b * oBatch;
  const int tid  = threadIdx.x;
  const int wid  = tid >> 5;
  const int lane = tid & 31;
  const int wm = (wid >> 1) * 32;
  const int wn = (wid & 1) * 64;
  const int l15 = lane & 15;
  const int akb = (lane < 16) ? 0 : 16;
  const int bkb = (lane < 16) ? 0 : 16;

  v8f zero = {};
  v8f acc[2][4];
  #pragma unroll
  for (int i = 0; i < 2; i++)
    #pragma unroll
    for (int j = 0; j < 4; j++) acc[i][j] = zero;

  for (int k0 = 0; k0 < K; k0 += 32) {
    // A tile: 64x32 bf16, direct b128 copy. 256 i4 / 128 thr = 2 each.
    #pragma unroll
    for (int j = 0; j < 2; j++) {
      int idx = tid + j * 128;            // 8-short chunk index
      int r = idx >> 2, c8 = idx & 3;     // 4 chunks per row
      *(i4*)&sW[r * 32 + c8 * 8] =
          *(const i4*)&Ab[(long)(m0 + r) * K + (k0 + c8 * 8)];
    }
    // X tile: 32(k) x 128(n) bf16 -> transposed [n][k].
    // unit = 2 k-rows x 8 n-cols; 256 units / 128 thr = 2 each.
    #pragma unroll
    for (int j = 0; j < 2; j++) {
      int u = tid + j * 128;
      int kk = (u & 15) * 2, nn = (u >> 4) * 8;
      Pack8 a, c;
      a.q = *(const i4*)&Xb[(long)(k0 + kk)     * N + (n0 + nn)];
      c.q = *(const i4*)&Xb[(long)(k0 + kk + 1) * N + (n0 + nn)];
      #pragma unroll
      for (int i = 0; i < 8; i++)
        *(unsigned int*)&sX[(nn + i) * 32 + kk] =
            (unsigned int)a.u[i] | ((unsigned int)c.u[i] << 16);
    }
    __syncthreads();
    FragAB a[2], bb[4];
    #pragma unroll
    for (int t = 0; t < 2; t++) {
      const unsigned short* ap = &sW[(wm + t * 16 + l15) * 32 + akb];
      a[t].q[0] = *(const i4*)ap;
      a[t].q[1] = *(const i4*)(ap + 8);
    }
    #pragma unroll
    for (int t = 0; t < 4; t++) {
      const unsigned short* bp = &sX[(wn + t * 16 + l15) * 32 + bkb];
      bb[t].q[0] = *(const i4*)bp;
      bb[t].q[1] = *(const i4*)(bp + 8);
    }
    #pragma unroll
    for (int tm = 0; tm < 2; tm++)
      #pragma unroll
      for (int tn = 0; tn < 4; tn++)
        acc[tm][tn] = __builtin_amdgcn_wmma_f32_16x16x32_bf16(
            false, a[tm].v, false, bb[tn].v, (short)0, acc[tm][tn], false, false);
    __syncthreads();
  }
  #pragma unroll
  for (int tm = 0; tm < 2; tm++) {
    int rb = m0 + wm + tm * 16 + ((lane < 16) ? 0 : 8);
    #pragma unroll
    for (int tn = 0; tn < 4; tn++) {
      int cc = n0 + wn + tn * 16 + l15;
      #pragma unroll
      for (int i = 0; i < 8; i++)
        Ob[(long)(rb + i) * N + cc] = acc[tm][tn][i];
    }
  }
}

// ---------------------------------------------------------------------------
// Row L2 norms over n=4096 of a bf16 [.,384,4096] tensor (batched stride).
// ---------------------------------------------------------------------------
__global__ __launch_bounds__(256) void k_rownorm(
    const unsigned short* __restrict__ src, float* __restrict__ norms,
    long batchStride)
{
  __shared__ float red[256];
  const int gr = blockIdx.x;
  const int b = gr / 384;
  const int r = gr - b * 384;
  const unsigned short* p = src + (long)b * batchStride + (long)r * 4096;
  float s = 0.0f;
  #pragma unroll
  for (int i = threadIdx.x; i < 512; i += 256) {   // 512 x 8 bf16 = 4096
    Pack8 d; d.q = ((const i4*)p)[i];
    #pragma unroll
    for (int j = 0; j < 8; j++) { float v = bf2f(d.u[j]); s += v * v; }
  }
  red[threadIdx.x] = s;
  __syncthreads();
  for (int st = 128; st > 0; st >>= 1) {
    if (threadIdx.x < st) red[threadIdx.x] += red[threadIdx.x + st];
    __syncthreads();
  }
  if (threadIdx.x == 0) norms[gr] = sqrtf(red[0]);
}

// ---------------------------------------------------------------------------
// Gram: S[b,h] += q_h[48 x chunk] @ k_h[48 x chunk]^T  (contraction over n).
// One wave per (chunk, head, batch); 3x3 WMMA tiles; fragments from global.
// ---------------------------------------------------------------------------
__global__ __launch_bounds__(32) void k_gram(
    const unsigned short* __restrict__ qb, const unsigned short* __restrict__ kvb,
    float* __restrict__ S)
{
  const int chunk = blockIdx.x;   // 16 chunks of 256 spatial positions
  const int h = blockIdx.y;
  const int b = blockIdx.z;
  const int lane = threadIdx.x;
  const int l15 = lane & 15;
  const unsigned short* qh = qb  + ((long)b * 384 + h * 48) * 4096;
  const unsigned short* kh = kvb + ((long)b * 768 + h * 48) * 4096;
  const int akb = (lane < 16) ? 0 : 16;
  const int bkb = (lane < 16) ? 0 : 16;

  v8f zero = {};
  v8f acc[3][3];
  #pragma unroll
  for (int i = 0; i < 3; i++)
    #pragma unroll
    for (int j = 0; j < 3; j++) acc[i][j] = zero;

  const int nbeg = chunk * 256;
  for (int n0 = nbeg; n0 < nbeg + 256; n0 += 32) {
    FragAB a[3], bb[3];
    #pragma unroll
    for (int t = 0; t < 3; t++) {
      const unsigned short* ap = qh + (long)(t * 16 + l15) * 4096 + n0 + akb;
      a[t].q[0] = *(const i4*)ap;
      a[t].q[1] = *(const i4*)(ap + 8);
      const unsigned short* bp = kh + (long)(t * 16 + l15) * 4096 + n0 + bkb;
      bb[t].q[0] = *(const i4*)bp;
      bb[t].q[1] = *(const i4*)(bp + 8);
    }
    #pragma unroll
    for (int tm = 0; tm < 3; tm++)
      #pragma unroll
      for (int tn = 0; tn < 3; tn++)
        acc[tm][tn] = __builtin_amdgcn_wmma_f32_16x16x32_bf16(
            false, a[tm].v, false, bb[tn].v, (short)0, acc[tm][tn], false, false);
  }
  float* Sb = S + (long)(b * 8 + h) * 2304;
  #pragma unroll
  for (int tm = 0; tm < 3; tm++) {
    int rb = tm * 16 + ((lane < 16) ? 0 : 8);
    #pragma unroll
    for (int tn = 0; tn < 3; tn++) {
      int cc = tn * 16 + l15;
      #pragma unroll
      for (int i = 0; i < 8; i++)
        atomicAdd(&Sb[(rb + i) * 48 + cc], acc[tm][tn][i]);
    }
  }
}

// ---------------------------------------------------------------------------
// attn = softmax(S / (||q_c||*||k_d||) * temp); then fold into
// M[b][o][h*48+d] = sum_c wproj[o][h*48+c] * attn[h][c][d]   (bf16 out)
// ---------------------------------------------------------------------------
__global__ __launch_bounds__(256) void k_attn_m(
    const float* __restrict__ S, const float* __restrict__ nq,
    const float* __restrict__ nk, const float* __restrict__ temp,
    const float* __restrict__ wproj, unsigned short* __restrict__ Mb)
{
  __shared__ float attn[48 * 48];
  const int h = blockIdx.x;
  const int b = blockIdx.y;
  const float* Sb  = S  + (long)(b * 8 + h) * 2304;
  const float* nqh = nq + b * 384 + h * 48;
  const float* nkh = nk + b * 384 + h * 48;
  const float t = temp[h];
  const int tid = threadIdx.x;

  if (tid < 48) {
    const float iq = 1.0f / fmaxf(nqh[tid], 1e-12f);
    float row[48];
    float mx = -3.402823466e38f;
    #pragma unroll
    for (int d = 0; d < 48; d++) {
      float v = Sb[tid * 48 + d] * iq / fmaxf(nkh[d], 1e-12f) * t;
      row[d] = v;
      mx = fmaxf(mx, v);
    }
    float s = 0.0f;
    #pragma unroll
    for (int d = 0; d < 48; d++) { row[d] = __expf(row[d] - mx); s += row[d]; }
    const float inv = 1.0f / s;
    #pragma unroll
    for (int d = 0; d < 48; d++) attn[tid * 48 + d] = row[d] * inv;
  }
  __syncthreads();

  for (int e = tid; e < 384 * 48; e += 256) {
    int o = e / 48, d = e - o * 48;
    const float* wrow = wproj + (long)o * 384 + h * 48;
    float s = 0.0f;
    #pragma unroll
    for (int c = 0; c < 48; c++) s += wrow[c] * attn[c * 48 + d];
    Mb[((long)b * 384 + o) * 384 + h * 48 + d] = f2bfh(s);
  }
}

// ---------------------------------------------------------------------------
extern "C" void kernel_launch(void* const* d_in, const int* in_sizes, int n_in,
                              void* d_out, int out_size, void* d_ws, size_t ws_size,
                              hipStream_t stream) {
  const float* x       = (const float*)d_in[0];   // [16,384,64,64]
  const float* query   = (const float*)d_in[1];   // [16,192,64,64]
  const float* w_kv    = (const float*)d_in[2];   // [768,384]
  const float* w_q     = (const float*)d_in[3];   // [384,192]
  const float* w_proj  = (const float*)d_in[4];   // [384,384]
  const float* temper  = (const float*)d_in[5];   // [8]
  float* out = (float*)d_out;

  const int B = 16, C = 384, HW = 4096, HD = 8, DC = 48, QC = 192;

  char* ws = (char*)d_ws;
  size_t off = 0;
  unsigned short* kvb = (unsigned short*)(ws + off); off += (size_t)B * 2 * C * HW * 2; // k|v bf16
  unsigned short* qb  = (unsigned short*)(ws + off); off += (size_t)B * C * HW * 2;     // q bf16
  float* Sbuf = (float*)(ws + off); off += (size_t)B * HD * DC * DC * 4;
  float* nq   = (float*)(ws + off); off += (size_t)B * C * 4;
  float* nk   = (float*)(ws + off); off += (size_t)B * C * 4;
  unsigned short* Mb = (unsigned short*)(ws + off); off += (size_t)B * C * C * 2;
  (void)ws_size; (void)in_sizes; (void)n_in; (void)out_size;

  // 1) kv = w_kv @ x   (k = rows 0..383, v = rows 384..767 of each batch)
  k_gemm_f32w_bf16out<<<dim3(HW / 128, 768 / 64, B), 128, 0, stream>>>(
      w_kv, x, kvb, 768, C, HW, (long)C * HW, (long)768 * HW);
  // 2) q = w_q @ query
  k_gemm_f32w_bf16out<<<dim3(HW / 128, C / 64, B), 128, 0, stream>>>(
      w_q, query, qb, C, QC, HW, (long)QC * HW, (long)C * HW);
  // 3) norms
  k_rownorm<<<B * C, 256, 0, stream>>>(qb, nq, (long)C * HW);
  k_rownorm<<<B * C, 256, 0, stream>>>(kvb, nk, (long)768 * HW);
  // 4) S = q @ k^T (per batch/head), split over 16 spatial chunks w/ atomics
  hipMemsetAsync(Sbuf, 0, (size_t)B * HD * DC * DC * 4, stream);
  k_gram<<<dim3(16, HD, B), 32, 0, stream>>>(qb, kvb, Sbuf);
  // 5) softmax + fold w_proj -> M[b] (384x384 bf16 per batch)
  k_attn_m<<<dim3(HD, B), 256, 0, stream>>>(Sbuf, nq, nk, temper, w_proj, Mb);
  // 6) y = M[b] @ v[b]
  k_gemm_bf16_f32out<<<dim3(HW / 128, C / 64, B), 128, 0, stream>>>(
      Mb, kvb + (size_t)C * HW, out, C, C, HW,
      (long)C * C, (long)768 * HW, (long)C * HW);
}
